// CausalSelfAttentionMuP_33947421508129
// MI455X (gfx1250) — matrix operand order repaired and checked by reading, not compiled, not run on hardware
//
#include <hip/hip_runtime.h>
#include <hip/hip_bf16.h>

// ---------------- problem constants ----------------
constexpr int Bq  = 4;
constexpr int Tq  = 2048;
constexpr int Cq  = 1024;
constexpr int Hq  = 16;
constexpr int HDq = 64;          // head dim
constexpr int BT  = Bq * Tq;     // 8192 rows
constexpr int N3  = 3 * Cq;      // 3072

typedef __attribute__((ext_vector_type(16))) _Float16 v16h;
typedef __attribute__((ext_vector_type(8)))  _Float16 v8h;
typedef __attribute__((ext_vector_type(8)))  float    v8f;

// ---------------- WMMA fragment helpers (16x16x32 f16; this code's contiguous k = 16h+s order, same for A and B) ----------------
// A (16x32, f16): lane L holds row M = L&15; lanes 0-15 carry K=0..15,
// lanes 16-31 carry K=16..31. Element order = VGPR0.lo,VGPR0.hi,...
__device__ __forceinline__ v16h frag_a16(const _Float16* rowptr, int lane) {
  const _Float16* p = rowptr + ((lane >> 4) * 16);
  v8h lo = *(const v8h*)(p);
  v8h hi = *(const v8h*)(p + 8);
  v16h r;
#pragma unroll
  for (int i = 0; i < 8; ++i) { r[i] = lo[i]; r[i + 8] = hi[i]; }
  return r;
}

// B (32x16, f16): lane L holds column N = L&15; lanes 0-15 carry K=0..15, lanes 16-31 K=16..31.
// colptr points at the K=0 element of this lane's column (column-contiguous storage).
__device__ __forceinline__ v16h frag_b16(const _Float16* colptr, int lane) {
  return *(const v16h*)(colptr + ((lane >> 4) * 16));
}

__device__ __forceinline__ v8f wmma_f16f32(v16h a, v16h b, v8f c) {
  return __builtin_amdgcn_wmma_f32_16x16x32_f16(false, a, false, b, (short)0, c, false, false);
}

// async DMA of 16 bytes global -> LDS (ASYNCcnt-tracked, gfx1250)
__device__ __forceinline__ void async_g2lds_b128(uint32_t lds_byte_addr, const _Float16* gptr) {
  asm volatile("global_load_async_to_lds_b128 %0, %1, off"
               :: "v"(lds_byte_addr), "v"(gptr) : "memory");
}
__device__ __forceinline__ void wait_asynccnt0() {
  asm volatile("s_wait_asynccnt 0x0" ::: "memory");
}

// ---------------- conversion / layout kernels ----------------
__global__ __launch_bounds__(256) void k_cvt_f32_f16(const float* __restrict__ in,
                                                     _Float16* __restrict__ out, int n) {
  int i = blockIdx.x * 256 + threadIdx.x;
  if (i < n) out[i] = (_Float16)in[i];
}

// in[rows][cols] f32 -> out[cols][rows] f16   (weights: row-major [K][N] -> col-major [N][K])
__global__ __launch_bounds__(256) void k_transpose_cvt(const float* __restrict__ in,
                                                       _Float16* __restrict__ out,
                                                       int rows, int cols) {
  int i = blockIdx.x * 256 + threadIdx.x;
  if (i < rows * cols) {
    int r = i / cols, c = i % cols;
    out[(size_t)c * rows + r] = (_Float16)in[i];
  }
}

// ---------------- shared GEMM body: 256 threads (8 waves), WG tile 128x64, K=1024 ----------
// B tile (64 cols x 32 k, f16, col-major) staged in LDS by async DMA, double-buffered.
// Each of the 8 waves computes a 16x64 slice with 4 chained WMMAs per k-step.
__device__ __forceinline__ void gemm_body_256(const _Float16* __restrict__ A,
                                              const _Float16* __restrict__ Bt,
                                              _Float16 (*ldsB)[64][32],  // [2][64][32]
                                              int arow, int nBase,
                                              int lane, int tid, v8f acc[4]) {
  constexpr int KD = Cq;
  const int col  = tid >> 2;          // 64 cols, 4 x 16B chunks per column
  const int koff = (tid & 3) * 8;     // f16 offset inside the 32-wide k slice
  // prologue: stage k-step 0 into buffer 0 (one b128 DMA per thread = 4 KB tile)
  {
    uint32_t lo = (uint32_t)(uintptr_t)(&ldsB[0][0][0]) + (uint32_t)tid * 16u;
    async_g2lds_b128(lo, Bt + (size_t)(nBase + col) * KD + koff);
  }
  for (int it = 0; it < KD / 32; ++it) {
    const int buf = it & 1;
    wait_asynccnt0();        // this thread's DMA chunk has landed in LDS
    __syncthreads();         // everyone's chunk has landed; prev buffer fully consumed
    if (it + 1 < KD / 32) {  // kick next k-step's DMA into the other buffer
      uint32_t lo = (uint32_t)(uintptr_t)(&ldsB[buf ^ 1][0][0]) + (uint32_t)tid * 16u;
      async_g2lds_b128(lo, Bt + (size_t)(nBase + col) * KD + (it + 1) * 32 + koff);
    }
    v16h a = frag_a16(A + (size_t)arow * KD + it * 32, lane);
#pragma unroll
    for (int nt = 0; nt < 4; ++nt) {
      const _Float16* colptr = &ldsB[buf][nt * 16 + (lane & 15)][0];
      acc[nt] = wmma_f16f32(a, frag_b16(colptr, lane), acc[nt]);
    }
  }
}

// ---------------- QKV GEMM: [BT,C] x [C,3C] + bias, scatter into head layouts ----------------
__global__ __launch_bounds__(256) void k_gemm_qkv(const _Float16* __restrict__ A,   // [BT][C]
                                                  const _Float16* __restrict__ Bt,  // [3C][C]
                                                  const float*    __restrict__ bias,// [3C]
                                                  _Float16* __restrict__ qh,  // [B,H,T,HD]
                                                  _Float16* __restrict__ kh,  // [B,H,T,HD]
                                                  _Float16* __restrict__ vt)  // [B,H,HD,T]
{
  __shared__ _Float16 ldsB[2][64][32];
  const int tid  = threadIdx.x;
  const int lane = tid & 31;
  const int wave = tid >> 5;
  const int nBase = blockIdx.x * 64;
  const int mBase = blockIdx.y * 128 + wave * 16;

  v8f acc[4] = {};
  gemm_body_256(A, Bt, ldsB, mBase + (lane & 15), nBase, lane, tid, acc);

  const int hi16 = lane >> 4;
#pragma unroll
  for (int nt = 0; nt < 4; ++nt) {
    const int n = nBase + nt * 16 + (lane & 15);
    const float bv = bias[n];
#pragma unroll
    for (int r = 0; r < 8; ++r) {
      const int m  = mBase + r + 8 * hi16;           // global token row
      const int t  = m & (Tq - 1);
      const int bb = m >> 11;                        // T = 2048
      const _Float16 hv = (_Float16)(acc[nt][r] + bv);
      if (n < Cq) {
        int hh = n >> 6, hd = n & 63;
        qh[(((size_t)(bb * Hq + hh)) * Tq + t) * HDq + hd] = hv;
      } else if (n < 2 * Cq) {
        int n2 = n - Cq; int hh = n2 >> 6, hd = n2 & 63;
        kh[(((size_t)(bb * Hq + hh)) * Tq + t) * HDq + hd] = hv;
      } else {
        int n2 = n - 2 * Cq; int hh = n2 >> 6, hd = n2 & 63;
        vt[(((size_t)(bb * Hq + hh)) * HDq + hd) * Tq + t] = hv;  // transposed V
      }
    }
  }
}

// ---------------- flash attention (causal, muP 1/HD scale) ----------------
// grid = (T/64, B*H); block = 128 (4 waves). Wave owns 16 query rows.
__global__ __launch_bounds__(128) void k_attn(const _Float16* __restrict__ qh,
                                              const _Float16* __restrict__ kh,
                                              const _Float16* __restrict__ vt,
                                              _Float16* __restrict__ y_att)  // [BT][C]
{
  __shared__ _Float16 ldsP[4][16][32];   // per-wave P staging (D-layout -> A-layout)

  const int lane = threadIdx.x & 31;
  const int wave = threadIdx.x >> 5;
  const int hi16 = lane >> 4;
  const int bh = blockIdx.y;
  const int b  = bh >> 4;
  const int h  = bh & 15;
  const int qb = blockIdx.x * 64 + wave * 16;   // first query row of this wave

  const _Float16* Q = qh + (size_t)bh * Tq * HDq;
  const _Float16* K = kh + (size_t)bh * Tq * HDq;
  const _Float16* V = vt + (size_t)bh * HDq * Tq;

  // Q fragments, K-dim = head dim (64) in two 32-chunks; resident whole kernel.
  const v16h qA0 = frag_a16(Q + (size_t)(qb + (lane & 15)) * HDq + 0,  lane);
  const v16h qA1 = frag_a16(Q + (size_t)(qb + (lane & 15)) * HDq + 32, lane);

  v8f O[4] = {};                 // 16 rows x 64 hd
  float mi[8], li[8];
#pragma unroll
  for (int r = 0; r < 8; ++r) { mi[r] = -__builtin_inff(); li[r] = 0.0f; }

  const float scale = 1.0f / (float)HDq;       // muP: 1/head_dim
  const int kmax = qb + 15;

  for (int kb = 0; kb <= kmax; kb += 32) {
    // S = Q*K^T : two 16x16 key tiles, chained over hd chunks
    v8f S0 = {}, S1 = {};
    {
      const _Float16* c0 = K + (size_t)(kb +      (lane & 15)) * HDq;
      const _Float16* c1 = K + (size_t)(kb + 16 + (lane & 15)) * HDq;
      S0 = wmma_f16f32(qA0, frag_b16(c0 + 0,  lane), S0);
      S0 = wmma_f16f32(qA1, frag_b16(c0 + 32, lane), S0);
      S1 = wmma_f16f32(qA0, frag_b16(c1 + 0,  lane), S1);
      S1 = wmma_f16f32(qA1, frag_b16(c1 + 32, lane), S1);
    }

    const int nc0 = kb + (lane & 15);
    const int nc1 = nc0 + 16;
#pragma unroll
    for (int r = 0; r < 8; ++r) {
      const int mrow = qb + r + 8 * hi16;
      float s0 = S0[r] * scale; if (nc0 > mrow) s0 = -__builtin_inff();
      float s1 = S1[r] * scale; if (nc1 > mrow) s1 = -__builtin_inff();
      // row max across the 16-lane group (both key tiles)
      float rm = fmaxf(s0, s1);
      rm = fmaxf(rm, __shfl_xor(rm, 1, 32));
      rm = fmaxf(rm, __shfl_xor(rm, 2, 32));
      rm = fmaxf(rm, __shfl_xor(rm, 4, 32));
      rm = fmaxf(rm, __shfl_xor(rm, 8, 32));
      const float mnew  = fmaxf(mi[r], rm);
      const float alpha = __expf(mi[r] - mnew);
      mi[r] = mnew;
      const float p0 = __expf(s0 - mnew);
      const float p1 = __expf(s1 - mnew);
      float rs = p0 + p1;
      rs += __shfl_xor(rs, 1, 32);
      rs += __shfl_xor(rs, 2, 32);
      rs += __shfl_xor(rs, 4, 32);
      rs += __shfl_xor(rs, 8, 32);
      li[r] = li[r] * alpha + rs;
#pragma unroll
      for (int j = 0; j < 4; ++j) O[j][r] *= alpha;
      ldsP[wave][r + 8 * hi16][(lane & 15)]      = (_Float16)p0;
      ldsP[wave][r + 8 * hi16][(lane & 15) + 16] = (_Float16)p1;
    }
    // DS ops from one wave complete in order; wait for our stores, then reload as A-frag.
    asm volatile("s_wait_dscnt 0" ::: "memory");
    const v16h pA = frag_a16(&ldsP[wave][lane & 15][0], lane);

    // O += P * V : K-dim = 32 keys, N = hd (V stored transposed -> contiguous keys)
#pragma unroll
    for (int j = 0; j < 4; ++j) {
      v16h vB = frag_b16(V + (size_t)(j * 16 + (lane & 15)) * Tq + kb, lane);
      O[j] = wmma_f16f32(pA, vB, O[j]);
    }
  }

  // epilogue: normalize and emit f16 rows for proj GEMM
#pragma unroll
  for (int r = 0; r < 8; ++r) {
    const float invl = 1.0f / li[r];
    const int qrow = qb + r + 8 * hi16;
    const size_t row = (size_t)(b * Tq + qrow) * Cq + h * HDq;
#pragma unroll
    for (int j = 0; j < 4; ++j)
      y_att[row + j * 16 + (lane & 15)] = (_Float16)(O[j][r] * invl);
  }
}

// ---------------- output projection: [BT,C] x [C,C] + bias -> f32 ----------------
__global__ __launch_bounds__(256) void k_gemm_proj(const _Float16* __restrict__ A,   // [BT][C]
                                                   const _Float16* __restrict__ Bt,  // [C][C]
                                                   const float*    __restrict__ bias,
                                                   float* __restrict__ out)          // [BT][C]
{
  __shared__ _Float16 ldsB[2][64][32];
  const int tid  = threadIdx.x;
  const int lane = tid & 31;
  const int wave = tid >> 5;
  const int nBase = blockIdx.x * 64;
  const int mBase = blockIdx.y * 128 + wave * 16;

  v8f acc[4] = {};
  gemm_body_256(A, Bt, ldsB, mBase + (lane & 15), nBase, lane, tid, acc);

  const int hi16 = lane >> 4;
#pragma unroll
  for (int nt = 0; nt < 4; ++nt) {
    const int n = nBase + nt * 16 + (lane & 15);
    const float bv = bias[n];
#pragma unroll
    for (int r = 0; r < 8; ++r) {
      const int m = mBase + r + 8 * hi16;
      out[(size_t)m * Cq + n] = acc[nt][r] + bv;
    }
  }
}

// ---------------- host launch ----------------
extern "C" void kernel_launch(void* const* d_in, const int* in_sizes, int n_in,
                              void* d_out, int out_size, void* d_ws, size_t ws_size,
                              hipStream_t stream) {
  const float* x      = (const float*)d_in[0];
  const float* w_qkv  = (const float*)d_in[1];
  const float* b_qkv  = (const float*)d_in[2];
  const float* w_proj = (const float*)d_in[3];
  const float* b_proj = (const float*)d_in[4];
  float* out = (float*)d_out;

  // workspace carve-up (~88 MB total)
  char* ws = (char*)d_ws;
  size_t off = 0;
  auto alloc = [&](size_t bytes) -> void* {
    void* p = ws + off; off += (bytes + 255) & ~(size_t)255; return p;
  };
  _Float16* x_h     = (_Float16*)alloc((size_t)BT * Cq * 2);   // 16 MB
  _Float16* wqkv_t  = (_Float16*)alloc((size_t)N3 * Cq * 2);   //  6 MB
  _Float16* wproj_t = (_Float16*)alloc((size_t)Cq * Cq * 2);   //  2 MB
  _Float16* q_h     = (_Float16*)alloc((size_t)BT * Cq * 2);   // 16 MB [B,H,T,HD]
  _Float16* k_h     = (_Float16*)alloc((size_t)BT * Cq * 2);   // 16 MB [B,H,T,HD]
  _Float16* v_t     = (_Float16*)alloc((size_t)BT * Cq * 2);   // 16 MB [B,H,HD,T]
  _Float16* y_att   = (_Float16*)alloc((size_t)BT * Cq * 2);   // 16 MB [BT,C]

  // 1) precision / layout conversion
  {
    int n = BT * Cq;
    k_cvt_f32_f16<<<(n + 255) / 256, 256, 0, stream>>>(x, x_h, n);
  }
  {
    int n = Cq * N3;
    k_transpose_cvt<<<(n + 255) / 256, 256, 0, stream>>>(w_qkv, wqkv_t, Cq, N3);
  }
  {
    int n = Cq * Cq;
    k_transpose_cvt<<<(n + 255) / 256, 256, 0, stream>>>(w_proj, wproj_t, Cq, Cq);
  }

  // 2) QKV projection (WMMA + async-LDS staged B), scatter to head-major q/k, transposed v
  k_gemm_qkv<<<dim3(N3 / 64, BT / 128), 256, 0, stream>>>(x_h, wqkv_t, b_qkv, q_h, k_h, v_t);

  // 3) causal flash attention (WMMA for QK^T and P*V)
  k_attn<<<dim3(Tq / 64, Bq * Hq), 128, 0, stream>>>(q_h, k_h, v_t, y_att);

  // 4) output projection (WMMA + async-LDS staged B) + bias, fp32 result
  k_gemm_proj<<<dim3(Cq / 64, BT / 128), 256, 0, stream>>>(y_att, wproj_t, b_proj, out);
}
